// PAM_40295383171514
// MI455X (gfx1250) — compile-verified, matrix operand order checked by reading, not run on hardware
//
#include <hip/hip_runtime.h>
#include <hip/hip_bf16.h>

// ---------------------------------------------------------------------------
// PAM stereo-attention block for MI455X (gfx1250, wave32, WMMA).
// All conv / 1x1-proj GEMM work runs through v_wmma_f32_16x16x32_f16.
// ---------------------------------------------------------------------------

typedef __attribute__((ext_vector_type(8)))  _Float16 v8h;
typedef __attribute__((ext_vector_type(16))) _Float16 v16h;
typedef __attribute__((ext_vector_type(8)))  float    v8f;

#define PAM_B  2
#define PAM_C  64
#define PAM_H  128
#define PAM_W  256
#define PAM_K  16
#define PAM_HW (PAM_H * PAM_W)   // 32768

// Build a 16-element f16 WMMA fragment from two contiguous 8-half (16B) runs.
// Matches the CDNA5 16-bit A/B layout: per lane, elements 0..7 = K(+0..7),
// elements 8..15 = K(+16..23); the lane's upper/lower half selects the K base.
static __device__ __forceinline__ v16h ld16(const _Float16* lo, const _Float16* hi) {
  v8h a = *(const v8h*)lo;
  v8h b = *(const v8h*)hi;
  v16h r;
#pragma unroll
  for (int i = 0; i < 8; ++i) { r[i] = a[i]; r[i + 8] = b[i]; }
  return r;
}

static __device__ __forceinline__ v8f wmma_f16(v16h a, v16h b, v8f c) {
  return __builtin_amdgcn_wmma_f32_16x16x32_f16(false, a, false, b, (short)0, c,
                                                false, false);
}

// ---------------------------------------------------------------------------
// Weight repack: f32 [co][ci][3][3] -> f16 [co][ (kh*3+kw)*64 + ci ]
// ---------------------------------------------------------------------------
__global__ void repack3x3(const float* __restrict__ in, _Float16* __restrict__ out) {
  int i = blockIdx.x * 256 + threadIdx.x;
  if (i >= PAM_C * PAM_C * 9) return;
  int kw = i % 3;
  int kh = (i / 3) % 3;
  int ci = (i / 9) % PAM_C;
  int co = i / (9 * PAM_C);
  out[co * 576 + (kh * 3 + kw) * PAM_C + ci] = (_Float16)in[i];
}

__global__ void castf16(const float* __restrict__ in, _Float16* __restrict__ out, int n) {
  int i = blockIdx.x * 256 + threadIdx.x;
  if (i < n) out[i] = (_Float16)in[i];
}

// ---------------------------------------------------------------------------
// 3x3 conv (pad=1, C=64->64) as implicit GEMM via WMMA.
// Block: 256 threads (8 waves). Block tile: 128 pixels (one H row segment) x 64 co.
// MODE 0: leaky_relu(y, 0.1)    (res-block conv1)
// MODE 1: y + residual (f32)    (res-block conv2)
// Input X: Tin NCHW. Output Y: f16 NCHW.
// ---------------------------------------------------------------------------
template <typename Tin, int MODE>
__global__ __launch_bounds__(256) void conv3x3(const Tin* __restrict__ X,
                                               const _Float16* __restrict__ Wk, // [64][576]
                                               const float* __restrict__ Res,   // f32 NCHW (MODE 1)
                                               _Float16* __restrict__ Y) {
  __shared__ _Float16 xs[3][PAM_W / 2 + 2][PAM_C]; // 3 x 130 x 64 f16, ci contiguous
  __shared__ _Float16 wsd[PAM_C][576];             // [co][k] f16

  const int tid = threadIdx.x;
  const int w0  = blockIdx.x * 128;
  const int h   = blockIdx.y;
  const int b   = blockIdx.z;
  const size_t imgBase = (size_t)b * PAM_C * PAM_HW;

  // Stage weights (4608 x 16B blocks).
  for (int bi = tid; bi < (PAM_C * 576) / 8; bi += 256) {
    int co = bi / 72, k8 = (bi % 72) * 8;
    *(v8h*)&wsd[co][k8] = *(const v8h*)&Wk[co * 576 + k8];
  }
  // Stage input patch with NCHW -> [row][col][ci] transpose + f16 convert.
  for (int idx = tid; idx < 3 * 130 * PAM_C; idx += 256) {
    int c  = idx % 130;
    int ci = (idx / 130) % PAM_C;
    int r  = idx / (130 * PAM_C);
    int hr = h + r - 1;
    int wc = w0 + c - 1;
    _Float16 v = (_Float16)0.f;
    if (hr >= 0 && hr < PAM_H && wc >= 0 && wc < PAM_W)
      v = (_Float16)(float)X[imgBase + (size_t)ci * PAM_HW + (size_t)hr * PAM_W + wc];
    xs[r][c][ci] = v;
  }
  __syncthreads();

  const int lane = tid & 31, wave = tid >> 5;
  const int mrow = lane & 15, lh = lane >> 4;
  const int pix  = wave * 16 + mrow; // this lane's M row (pixel within block tile)

  v8f acc0 = {}, acc1 = {}, acc2 = {}, acc3 = {};
#pragma unroll
  for (int kh = 0; kh < 3; ++kh) {
#pragma unroll
    for (int kw = 0; kw < 3; ++kw) {
#pragma unroll
      for (int ch = 0; ch < 2; ++ch) { // two 32-wide ci chunks
        const int cb = ch * 32 + lh * 8;
        const int kb = (kh * 3 + kw) * PAM_C + cb;
        v16h a  = ld16(&xs[kh][pix + kw][cb], &xs[kh][pix + kw][cb + 16]);
        v16h b0 = ld16(&wsd[mrow +  0][kb], &wsd[mrow +  0][kb + 16]);
        v16h b1 = ld16(&wsd[mrow + 16][kb], &wsd[mrow + 16][kb + 16]);
        v16h b2 = ld16(&wsd[mrow + 32][kb], &wsd[mrow + 32][kb + 16]);
        v16h b3 = ld16(&wsd[mrow + 48][kb], &wsd[mrow + 48][kb + 16]);
        acc0 = wmma_f16(a, b0, acc0);
        acc1 = wmma_f16(a, b1, acc1);
        acc2 = wmma_f16(a, b2, acc2);
        acc3 = wmma_f16(a, b3, acc3);
      }
    }
  }

  // Epilogue: C/D layout -> lane holds N=mrow, rows M = lh*8 + j.
  const int prow = wave * 16 + lh * 8;
  v8f accs[4] = {acc0, acc1, acc2, acc3};
#pragma unroll
  for (int nt = 0; nt < 4; ++nt) {
    const int co = nt * 16 + mrow;
    const size_t off = imgBase + (size_t)co * PAM_HW + (size_t)h * PAM_W + (w0 + prow);
    v8f v = accs[nt];
    if (MODE == 0) {
#pragma unroll
      for (int j = 0; j < 8; ++j) v[j] = (v[j] > 0.f) ? v[j] : 0.1f * v[j];
    } else {
      const float4* rp = reinterpret_cast<const float4*>(&Res[off]);
      float4 r0 = rp[0], r1 = rp[1];
      v[0] += r0.x; v[1] += r0.y; v[2] += r0.z; v[3] += r0.w;
      v[4] += r1.x; v[5] += r1.y; v[6] += r1.z; v[7] += r1.w;
    }
    v8h o;
#pragma unroll
    for (int j = 0; j < 8; ++j) o[j] = (_Float16)v[j];
    *(v8h*)&Y[off] = o;
  }
}

// ---------------------------------------------------------------------------
// 1x1 conv (C=64->64) GEMM via WMMA, with bias.
// layoutQ=1: Out is [b][p][64] f16 (attention Q layout)
// layoutQ=0: Out is [b][64][p] f16 (gatherable S/R layout)
// ---------------------------------------------------------------------------
__global__ __launch_bounds__(256) void gemm1x1(const _Float16* __restrict__ In, // [B][64][HW]
                                               const _Float16* __restrict__ Wm, // [64][64]
                                               const float* __restrict__ bias,  // [64]
                                               _Float16* __restrict__ Out,
                                               int layoutQ) {
  __shared__ _Float16 as[128][PAM_C];
  __shared__ _Float16 wl[PAM_C][PAM_C];

  const int tid = threadIdx.x;
  const int p0  = blockIdx.x * 128;
  const int b   = blockIdx.y;

  for (int bi = tid; bi < (PAM_C * PAM_C) / 8; bi += 256) {
    int co = bi / 8, k8 = (bi % 8) * 8;
    *(v8h*)&wl[co][k8] = *(const v8h*)&Wm[co * PAM_C + k8];
  }
  for (int idx = tid; idx < 128 * PAM_C; idx += 256) {
    int p = idx % 128, ci = idx / 128;
    as[p][ci] = In[((size_t)b * PAM_C + ci) * PAM_HW + p0 + p];
  }
  __syncthreads();

  const int lane = tid & 31, wave = tid >> 5;
  const int mrow = lane & 15, lh = lane >> 4;
  const int pix  = wave * 16 + mrow;

  v8f acc0 = {}, acc1 = {}, acc2 = {}, acc3 = {};
#pragma unroll
  for (int kc = 0; kc < 2; ++kc) {
    const int kb = kc * 32 + lh * 8;
    v16h a  = ld16(&as[pix][kb], &as[pix][kb + 16]);
    v16h b0 = ld16(&wl[mrow +  0][kb], &wl[mrow +  0][kb + 16]);
    v16h b1 = ld16(&wl[mrow + 16][kb], &wl[mrow + 16][kb + 16]);
    v16h b2 = ld16(&wl[mrow + 32][kb], &wl[mrow + 32][kb + 16]);
    v16h b3 = ld16(&wl[mrow + 48][kb], &wl[mrow + 48][kb + 16]);
    acc0 = wmma_f16(a, b0, acc0);
    acc1 = wmma_f16(a, b1, acc1);
    acc2 = wmma_f16(a, b2, acc2);
    acc3 = wmma_f16(a, b3, acc3);
  }

  const int prow = wave * 16 + lh * 8;
  v8f accs[4] = {acc0, acc1, acc2, acc3};
#pragma unroll
  for (int nt = 0; nt < 4; ++nt) {
    const int co = nt * 16 + mrow;
    const float bv = bias[co];
    v8f v = accs[nt];
    if (layoutQ) {
#pragma unroll
      for (int j = 0; j < 8; ++j)
        Out[((size_t)b * PAM_HW + p0 + prow + j) * PAM_C + co] = (_Float16)(v[j] + bv);
    } else {
      v8h o;
#pragma unroll
      for (int j = 0; j < 8; ++j) o[j] = (_Float16)(v[j] + bv);
      *(v8h*)&Out[((size_t)b * PAM_C + co) * PAM_HW + p0 + prow] = o;
    }
  }
}

// ---------------------------------------------------------------------------
// Attention: per pixel, gather 16 candidates from S/R, score = Q . Key,
// softmax over K=16, out = M . Val. One wave (32 lanes) per pixel.
// ---------------------------------------------------------------------------
__global__ __launch_bounds__(256) void attention(const _Float16* __restrict__ Qf, // [B][HW][64]
                                                 const _Float16* __restrict__ Sf, // [B][64][HW]
                                                 const _Float16* __restrict__ Rf, // [B][64][HW]
                                                 const int* __restrict__ xxs,     // [B][HW][16]
                                                 const int* __restrict__ yys,
                                                 float* __restrict__ Mout,        // [B][HW][16]
                                                 _Float16* __restrict__ Batt) {   // [B][HW][64]
  const int tid  = threadIdx.x;
  const int lane = tid & 31, wv = tid >> 5;
  const long gp  = (long)blockIdx.x * 8 + wv; // global pixel index in [0, B*HW)
  const int b    = (int)(gp >> 15);
  __shared__ float qsh[8][PAM_C];

  qsh[wv][lane]      = (float)Qf[gp * PAM_C + lane];
  qsh[wv][lane + 32] = (float)Qf[gp * PAM_C + lane + 32];
  __syncthreads();

  // Candidate index for this lane's k (replicated across lane halves).
  const int k  = lane & 15;
  const int fk = xxs[gp * PAM_K + k] * PAM_W + yys[gp * PAM_K + k];

  // Score: lanes 0-15 reduce c in [0,32), lanes 16-31 reduce c in [32,64).
  const _Float16* Sb = Sf + (size_t)b * PAM_C * PAM_HW;
  const int c0 = (lane >> 4) * 32;
  float part = 0.f;
#pragma unroll 8
  for (int c = 0; c < 32; ++c)
    part += qsh[wv][c0 + c] * (float)Sb[(size_t)(c0 + c) * PAM_HW + fk];
  float sc = part + __shfl_xor(part, 16, 32); // full dot, replicated in both halves

  // Softmax over the 16 k-values (replicated halves keep xor masks 1,2,4,8 valid).
  float mx = sc;
#pragma unroll
  for (int m = 8; m >= 1; m >>= 1) mx = fmaxf(mx, __shfl_xor(mx, m, 32));
  float e = __expf(sc - mx);
  float s = e;
#pragma unroll
  for (int m = 8; m >= 1; m >>= 1) s += __shfl_xor(s, m, 32);
  const float mval = e / s;

  if (lane < 16) Mout[gp * PAM_K + k] = mval;

  // out_c = sum_k M_k * R[c][idx_k]; lane owns channels c = lane, lane+32.
  const _Float16* Rb = Rf + (size_t)b * PAM_C * PAM_HW;
  float a0 = 0.f, a1 = 0.f;
#pragma unroll
  for (int kk = 0; kk < 16; ++kk) {
    float mk  = __shfl(mval, kk, 32);
    int   fkk = __shfl(fk, kk, 32);
    a0 += mk * (float)Rb[(size_t)lane * PAM_HW + fkk];
    a1 += mk * (float)Rb[(size_t)(lane + 32) * PAM_HW + fkk];
  }
  Batt[gp * PAM_C + lane]      = (_Float16)a0;
  Batt[gp * PAM_C + lane + 32] = (_Float16)a1;
}

// ---------------------------------------------------------------------------
// Fusion 1x1 conv: out = fus_w . concat(Batt, x_left) + fus_b. K = 128.
// K chunks 0-1 come straight from Batt (already [p][c] f16 in global);
// chunks 2-3 come from x_left staged/converted in LDS.
// ---------------------------------------------------------------------------
__global__ __launch_bounds__(256) void fuse1x1(const _Float16* __restrict__ Batt, // [B][HW][64]
                                               const float* __restrict__ XL,      // [B][64][HW]
                                               const _Float16* __restrict__ Wf,   // [64][128]
                                               const float* __restrict__ bias,    // [64]
                                               float* __restrict__ Out) {         // [B][64][HW]
  __shared__ _Float16 xls[128][PAM_C];
  __shared__ _Float16 wfs[PAM_C][128];

  const int tid = threadIdx.x;
  const int p0  = blockIdx.x * 128;
  const int b   = blockIdx.y;

  for (int bi = tid; bi < (PAM_C * 128) / 8; bi += 256) {
    int co = bi / 16, k8 = (bi % 16) * 8;
    *(v8h*)&wfs[co][k8] = *(const v8h*)&Wf[co * 128 + k8];
  }
  for (int idx = tid; idx < 128 * PAM_C; idx += 256) {
    int p = idx % 128, ci = idx / 128;
    xls[p][ci] = (_Float16)XL[((size_t)b * PAM_C + ci) * PAM_HW + p0 + p];
  }
  __syncthreads();

  const int lane = tid & 31, wave = tid >> 5;
  const int mrow = lane & 15, lh = lane >> 4;
  const int pix  = wave * 16 + mrow;
  const _Float16* arow = &Batt[((size_t)b * PAM_HW + p0 + pix) * PAM_C];

  v8f acc0 = {}, acc1 = {}, acc2 = {}, acc3 = {};
#pragma unroll
  for (int kc = 0; kc < 4; ++kc) {
    const int kb = (kc & 1) * 32 + lh * 8;
    v16h a = (kc < 2) ? ld16(arow + kb, arow + kb + 16)
                      : ld16(&xls[pix][kb], &xls[pix][kb + 16]);
    const int wk = kc * 32 + lh * 8;
    v16h b0 = ld16(&wfs[mrow +  0][wk], &wfs[mrow +  0][wk + 16]);
    v16h b1 = ld16(&wfs[mrow + 16][wk], &wfs[mrow + 16][wk + 16]);
    v16h b2 = ld16(&wfs[mrow + 32][wk], &wfs[mrow + 32][wk + 16]);
    v16h b3 = ld16(&wfs[mrow + 48][wk], &wfs[mrow + 48][wk + 16]);
    acc0 = wmma_f16(a, b0, acc0);
    acc1 = wmma_f16(a, b1, acc1);
    acc2 = wmma_f16(a, b2, acc2);
    acc3 = wmma_f16(a, b3, acc3);
  }

  const int prow = wave * 16 + lh * 8;
  v8f accs[4] = {acc0, acc1, acc2, acc3};
#pragma unroll
  for (int nt = 0; nt < 4; ++nt) {
    const int co = nt * 16 + mrow;
    const float bv = bias[co];
    v8f v = accs[nt];
    float4* op = reinterpret_cast<float4*>(&Out[((size_t)b * PAM_C + co) * PAM_HW + p0 + prow]);
    op[0] = make_float4(v[0] + bv, v[1] + bv, v[2] + bv, v[3] + bv);
    op[1] = make_float4(v[4] + bv, v[5] + bv, v[6] + bv, v[7] + bv);
  }
}

// ---------------------------------------------------------------------------
// Host launcher
// ---------------------------------------------------------------------------
extern "C" void kernel_launch(void* const* d_in, const int* in_sizes, int n_in,
                              void* d_out, int out_size, void* d_ws, size_t ws_size,
                              hipStream_t stream) {
  const float* x_left  = (const float*)d_in[0];
  const float* x_right = (const float*)d_in[1];
  const int*   xxs     = (const int*)d_in[2];
  const int*   yys     = (const int*)d_in[3];
  const float* rb_w1   = (const float*)d_in[5];
  const float* rb_w2   = (const float*)d_in[6];
  const float* b1_w    = (const float*)d_in[7];
  const float* b1_b    = (const float*)d_in[8];
  const float* b2_w    = (const float*)d_in[9];
  const float* b2_b    = (const float*)d_in[10];
  const float* b3_w    = (const float*)d_in[11];
  const float* b3_b    = (const float*)d_in[12];
  const float* fus_w   = (const float*)d_in[13];
  const float* fus_b   = (const float*)d_in[14];

  float* out_main = (float*)d_out;                                   // [B][C][H][W]
  float* out_M    = out_main + (size_t)PAM_B * PAM_C * PAM_HW;       // [B][HW][1][K]

  // Bump allocator over workspace (f16 regions, 256B aligned).
  uintptr_t cur = (uintptr_t)d_ws;
  auto alloc_h = [&](size_t nelem) -> _Float16* {
    cur = (cur + 255) & ~(uintptr_t)255;
    _Float16* p = (_Float16*)cur;
    cur += nelem * sizeof(_Float16);
    return p;
  };
  const size_t IMG = (size_t)PAM_B * PAM_C * PAM_HW; // 4,194,304 elems
  _Float16* wk1  = alloc_h(PAM_C * 576);
  _Float16* wk2  = alloc_h(PAM_C * 576);
  _Float16* wq   = alloc_h(PAM_C * PAM_C);
  _Float16* wsm  = alloc_h(PAM_C * PAM_C);
  _Float16* wr   = alloc_h(PAM_C * PAM_C);
  _Float16* wf   = alloc_h(PAM_C * 2 * PAM_C);
  _Float16* hbuf = alloc_h(IMG);   // conv1 intermediate (reused left then right)
  _Float16* bufl = alloc_h(IMG);
  _Float16* bufr = alloc_h(IMG);
  _Float16* Qf   = alloc_h(IMG);
  _Float16* Sf   = alloc_h(IMG);
  _Float16* Rf   = alloc_h(IMG);
  _Float16* Batt = alloc_h(IMG);
  (void)ws_size; (void)n_in; (void)in_sizes; (void)out_size;

  // 1) Weight repacks / casts to f16.
  repack3x3<<<144, 256, 0, stream>>>(rb_w1, wk1);
  repack3x3<<<144, 256, 0, stream>>>(rb_w2, wk2);
  castf16<<<16, 256, 0, stream>>>(b1_w, wq, PAM_C * PAM_C);
  castf16<<<16, 256, 0, stream>>>(b2_w, wsm, PAM_C * PAM_C);
  castf16<<<16, 256, 0, stream>>>(b3_w, wr, PAM_C * PAM_C);
  castf16<<<32, 256, 0, stream>>>(fus_w, wf, PAM_C * 2 * PAM_C);

  // 2) Shared res-block on both images (WMMA implicit GEMM).
  dim3 cg(PAM_W / 128, PAM_H, PAM_B);
  conv3x3<float, 0><<<cg, 256, 0, stream>>>(x_left, wk1, x_left, hbuf);
  conv3x3<_Float16, 1><<<cg, 256, 0, stream>>>(hbuf, wk2, x_left, bufl);
  conv3x3<float, 0><<<cg, 256, 0, stream>>>(x_right, wk1, x_right, hbuf);
  conv3x3<_Float16, 1><<<cg, 256, 0, stream>>>(hbuf, wk2, x_right, bufr);

  // 3) Q/S/R projections (WMMA GEMM, K=64).
  dim3 gg(PAM_HW / 128, PAM_B);
  gemm1x1<<<gg, 256, 0, stream>>>(bufl, wq, b1_b, Qf, 1);
  gemm1x1<<<gg, 256, 0, stream>>>(bufr, wsm, b2_b, Sf, 0);
  gemm1x1<<<gg, 256, 0, stream>>>(bufr, wr, b3_b, Rf, 0);

  // 4) Gather + softmax attention (one wave per pixel).
  attention<<<(PAM_B * PAM_HW) / 8, 256, 0, stream>>>(Qf, Sf, Rf, xxs, yys, out_M, Batt);

  // 5) Fusion 1x1 over concat(Batt, x_left) (WMMA GEMM, K=128) -> f32 out.
  fuse1x1<<<gg, 256, 0, stream>>>(Batt, x_left, wf, fus_b, out_main);
}
